// ESN_20426864460548
// MI455X (gfx1250) — matrix operand order repaired and checked by reading, not compiled, not run on hardware
//
#include <hip/hip_runtime.h>
#include <hip/hip_bf16.h>
#include <math.h>

// ---------------- CDNA5 WMMA types ----------------
typedef __bf16 bf16_t;
typedef __attribute__((ext_vector_type(16))) __bf16 v16bf;
typedef __attribute__((ext_vector_type(8)))  float  v8f;
typedef __attribute__((ext_vector_type(4)))  int    i128;   // POD 16-byte chunk

union FragBF { v16bf v; i128 q[2]; };

#define B_   16
#define T_   2048
#define I_   32
#define N_   2048
#define O_   32
#define G_   64      // persistent workgroups (each owns 32 columns of h)
#define WPB  128     // 4 wave32 per workgroup
#define LEAK 0.9f

// LDS layout (bytes) -- dynamic LDS only, so region offsets == LDS byte addresses
#define OFF_H    0                         // h_t, row-major [16][2048] bf16 = 65536
#define OFF_ATF  65536                     // A^T slice in B-fragment lane order: 64*2*32*16*2 = 131072
#define OFF_WINF (65536 + 131072)          // Win slice fragments: 2*32*16*2 = 2048
#define OFF_REDC (65536 + 131072 + 2048)   // split-K reduction: 2*16*16*4   = 2048
#define LDS_BYTES (65536 + 131072 + 2048 + 2048)

__global__ void esn_init(char* ws) {
  unsigned* bar = (unsigned*)(ws + (size_t)2 * B_ * N_ * 2);
  bar[0] = 0u;   // arrival counter
  bar[1] = 0u;   // generation
}

__global__ void __launch_bounds__(WPB, 1)
esn_kernel(const float* __restrict__ input, const float* __restrict__ Win,
           const float* __restrict__ A, const float* __restrict__ Wout,
           float* __restrict__ out, int Tkeep, char* __restrict__ ws)
{
  extern __shared__ char smem[];
  bf16_t* hLDS    = (bf16_t*)(smem + OFF_H);
  bf16_t* atFrag  = (bf16_t*)(smem + OFF_ATF);
  bf16_t* winFrag = (bf16_t*)(smem + OFF_WINF);
  float*  redC    = (float*)(smem + OFF_REDC);

  bf16_t* hbuf0 = (bf16_t*)ws;
  bf16_t* hbuf1 = hbuf0 + (size_t)B_ * N_;
  unsigned* barCnt = (unsigned*)(ws + (size_t)2 * B_ * N_ * 2);
  unsigned* barGen = barCnt + 1;

  const int tid    = threadIdx.x;
  const int wave   = tid >> 5;
  const int lane   = tid & 31;
  const int half   = (lane >> 4) & 1;
  const int lm     = lane & 15;
  const int ntile  = wave & 1;     // which 16-col N tile of this WG's 32-col slice
  const int kchunk = wave >> 1;    // split-K half: [0,1024) / [1024,2048)
  const int g      = blockIdx.x;
  const int nG     = g * 32 + ntile * 16 + lm;   // global reservoir column of this lane
  const int washm1 = T_ - Tkeep;                 // washout_len - 1

  // ---- init: h_0 = 0 in LDS ----
  {
    i128 z = {0, 0, 0, 0};
    i128* hq = (i128*)hLDS;                       // 65536/16 = 4096 chunks
    for (int i = tid; i < 4096; i += WPB) hq[i] = z;
  }
  // ---- init: stage A^T slice into LDS, pre-swizzled to B-fragment lane order ----
  // B 32x16 fragment: lane L -> n = L%16 col, element e -> K = 16*half + e
  for (int ks = kchunk * 32; ks < kchunk * 32 + 32; ++ks) {
    const float* arow = A + (size_t)nG * N_ + ks * 32 + half * 16;   // A[n][k..k+15]
    bf16_t* dst = atFrag + ((size_t)(ks * 2 + ntile) * 32 + lane) * 16;
    #pragma unroll
    for (int e = 0; e < 16; ++e) dst[e] = (bf16_t)arow[e];
  }
  // ---- init: Win slice fragments (used only by kchunk==0 waves) ----
  if (kchunk == 0) {
    const float* wrow = Win + (size_t)nG * I_ + half * 16;
    bf16_t* dst = winFrag + ((size_t)(ntile * 32) + lane) * 16;
    #pragma unroll
    for (int e = 0; e < 16; ++e) dst[e] = (bf16_t)wrow[e];
  }
  __syncthreads();

  float hm[8];                         // fp32 master copy of this lane's h slice
  #pragma unroll
  for (int r = 0; r < 8; ++r) hm[r] = 0.0f;

  for (int t = 0; t < T_; ++t) {
    // ---- (1) C = x_t * Win^T (tile)  then  C += h_t * A^T over this wave's K half
    v8f acc = {};
    if (kchunk == 0) {
      FragBF fa, fb;
      // A 16x32 fragment of x_t: lane m = lm, K(e) = e + 16*half
      const float* ip = input + (((size_t)lm * T_ + t) * I_) + half * 16;
      #pragma unroll
      for (int e = 0; e < 8; ++e) {
        fa.v[e]     = (bf16_t)ip[e];
        fa.v[8 + e] = (bf16_t)ip[8 + e];
      }
      const i128* wp = (const i128*)(winFrag + ((size_t)(ntile * 32) + lane) * 16);
      fb.q[0] = wp[0]; fb.q[1] = wp[1];
      acc = __builtin_amdgcn_wmma_f32_16x16x32_bf16(false, fa.v, false, fb.v,
                                                    (short)0, acc, false, false);
      if (t + 1 < T_) __builtin_prefetch(ip + I_, 0, 3);   // next step's x row
    }
    #pragma unroll 4
    for (int aks = kchunk * 32; aks < kchunk * 32 + 32; ++aks) {
      FragBF a, b;
      const int kb = aks * 32 + half * 16;
      const i128* hp = (const i128*)(hLDS + (size_t)lm * N_ + kb);
      a.q[0] = hp[0];      // halfs K = kb .. kb+7
      a.q[1] = hp[1];      // halfs K = kb+8 .. kb+15
      const i128* bp = (const i128*)(atFrag + ((size_t)(aks * 2 + ntile) * 32 + lane) * 16);
      b.q[0] = bp[0]; b.q[1] = bp[1];
      acc = __builtin_amdgcn_wmma_f32_16x16x32_bf16(false, a.v, false, b.v,
                                                    (short)0, acc, false, false);
    }
    // ---- split-K reduction + activation + leaky update + publish h slice
    __syncthreads();
    if (kchunk == 1) {
      #pragma unroll
      for (int r = 0; r < 8; ++r) redC[ntile * 256 + (r + 8 * half) * 16 + lm] = acc[r];
    }
    __syncthreads();
    if (kchunk == 0) {
      bf16_t* hdst = ((t + 1) & 1) ? hbuf1 : hbuf0;
      #pragma unroll
      for (int r = 0; r < 8; ++r) {
        float v  = acc[r] + redC[ntile * 256 + (r + 8 * half) * 16 + lm];
        float hn = (1.0f - LEAK) * hm[r] + LEAK * tanhf(v);
        hm[r] = hn;
        hdst[(size_t)(r + 8 * half) * N_ + nG] = (bf16_t)hn;
      }
      __threadfence();
    }
    // ---- (2) grid-wide sense barrier over G_ workgroups
    __syncthreads();
    if (tid == 0) {
      unsigned gen = __hip_atomic_load(barGen, __ATOMIC_ACQUIRE, __HIP_MEMORY_SCOPE_AGENT);
      unsigned n = __hip_atomic_fetch_add(barCnt, 1u, __ATOMIC_ACQ_REL, __HIP_MEMORY_SCOPE_AGENT);
      if (n == G_ - 1u) {
        __hip_atomic_store(barCnt, 0u, __ATOMIC_RELAXED, __HIP_MEMORY_SCOPE_AGENT);
        __hip_atomic_fetch_add(barGen, 1u, __ATOMIC_ACQ_REL, __HIP_MEMORY_SCOPE_AGENT);
      } else {
        while (__hip_atomic_load(barGen, __ATOMIC_ACQUIRE, __HIP_MEMORY_SCOPE_AGENT) == gen)
          __builtin_amdgcn_s_sleep(2);
      }
    }
    __syncthreads();
    __threadfence();
    // ---- (3) pull full h_{t+1} (64 KB bf16, L2-hot) into LDS via the CDNA5
    //      async global->LDS direct path (no VGPR staging, tracked by ASYNCcnt).
    //      Dynamic-LDS-only kernel => hLDS region lives at LDS byte offset OFF_H(=0).
    {
      const bf16_t* srcbuf = ((t + 1) & 1) ? hbuf1 : hbuf0;
      #pragma unroll
      for (int i = 0; i < 32; ++i) {
        unsigned off     = (unsigned)((tid + i * WPB) * 16);  // byte offset in 64KB buffer
        unsigned ldsAddr = (unsigned)OFF_H + off;             // LDS byte address
        // GVS mode: vdst = LDS byte address, vaddr = 32-bit global byte offset,
        // saddr = 64-bit base. One instruction moves 512B per wave, LDS-direct.
        asm volatile("global_load_async_to_lds_b128 %0, %1, %2"
                     :: "v"(ldsAddr), "v"(off), "s"(srcbuf)
                     : "memory");
      }
      asm volatile("s_wait_asynccnt 0x0" ::: "memory");
    }
    __syncthreads();
    // ---- (4) round-robin fused readout: this WG projects h_{t+1} -> out row t
    if (t >= washm1 && (t % G_) == g) {
      v8f oacc = {};
      #pragma unroll 4
      for (int aks = kchunk * 32; aks < kchunk * 32 + 32; ++aks) {
        FragBF a, b;
        const int kb = aks * 32 + half * 16;
        const i128* hp = (const i128*)(hLDS + (size_t)lm * N_ + kb);
        a.q[0] = hp[0];
        a.q[1] = hp[1];
        const float* wr = Wout + (size_t)(ntile * 16 + lm) * N_ + aks * 32 + half * 16;
        #pragma unroll
        for (int e = 0; e < 16; ++e) b.v[e] = (bf16_t)wr[e];
        oacc = __builtin_amdgcn_wmma_f32_16x16x32_bf16(false, a.v, false, b.v,
                                                       (short)0, oacc, false, false);
      }
      __syncthreads();
      if (kchunk == 1) {
        #pragma unroll
        for (int r = 0; r < 8; ++r) redC[ntile * 256 + (r + 8 * half) * 16 + lm] = oacc[r];
      }
      __syncthreads();
      if (kchunk == 0) {
        #pragma unroll
        for (int r = 0; r < 8; ++r) {
          float v = oacc[r] + redC[ntile * 256 + (r + 8 * half) * 16 + lm];
          out[((size_t)(r + 8 * half) * Tkeep + (t - washm1)) * O_ + ntile * 16 + lm] = v;
        }
      }
      __syncthreads();
    }
  }
}

extern "C" void kernel_launch(void* const* d_in, const int* in_sizes, int n_in,
                              void* d_out, int out_size, void* d_ws, size_t ws_size,
                              hipStream_t stream) {
  const float* input = (const float*)d_in[0];   // (16, 2048, 32)
  const float* Win   = (const float*)d_in[1];   // (2048, 32)
  const float* A     = (const float*)d_in[2];   // (2048, 2048)
  const float* Wout  = (const float*)d_in[3];   // (32, 2048)
  float* out = (float*)d_out;                   // (16, Tkeep, 32)
  int Tkeep = out_size / (B_ * O_);             // = T - washout + 1

  hipFuncSetAttribute(reinterpret_cast<const void*>(esn_kernel),
                      hipFuncAttributeMaxDynamicSharedMemorySize, LDS_BYTES);

  esn_init<<<1, 1, 0, stream>>>((char*)d_ws);
  esn_kernel<<<G_, WPB, LDS_BYTES, stream>>>(input, Win, A, Wout, out, Tkeep, (char*)d_ws);
}
